// GeometricFlow_74732430950680
// MI455X (gfx1250) — compile-verified, matrix operand order checked by reading, not run on hardware
//
#include <hip/hip_runtime.h>

typedef __attribute__((ext_vector_type(16))) _Float16 v16h;
typedef __attribute__((ext_vector_type(8)))  float    v8f;
typedef __attribute__((ext_vector_type(4)))  int      v4i;

// ---- workspace layout (float32 region, offsets in floats) ----
static constexpr size_t OFF_METRIC = 0;        // 32*32*32  = 32768
static constexpr size_t OFF_PTST   = 32768;    // 32*256    = 8192
static constexpr size_t OFF_G1     = 40960;    // 32*32*256 = 262144
static constexpr size_t OFF_G2     = 303104;   // 32*32*256 = 262144
static constexpr size_t OFF_HSUM   = 565248;   // 32*256    = 8192
static constexpr size_t OFF_S      = 573440;   // 32*1024   = 32768
static constexpr size_t F32_WS_FLOATS = 606208;  // then f16 region (32B-aligned)

// Fragment index helpers (16-bit A/B layout, wave32):
//   element e of lane L maps to k = (e/8)*16 + (L/16)*8 + (e%8), row/col = L%16 (+16*mt tile)

// K0: pre-pack Wr1 chris-slice and Wr2 into per-lane-contiguous WMMA B-fragment
// layout so each fragment load is 2x global_load_b128 instead of 16x u16.
__global__ void k0_pack(const float* __restrict__ Wr1, const float* __restrict__ Wr2,
                        _Float16* __restrict__ wr1cP, _Float16* __restrict__ wr2P) {
  int idx = blockIdx.x * 256 + threadIdx.x;            // 0..262143 covers wr2P exactly
  {
    int e = idx & 15, lane = (idx >> 4) & 31, tile = idx >> 9;   // tile = kt*64+nt
    int kt = tile >> 6, nt = tile & 63;
    int kk = kt * 32 + ((e >> 3) << 4) + ((lane >> 4) << 3) + (e & 7);
    wr2P[idx] = (_Float16)Wr2[kk * 1024 + nt * 16 + (lane & 15)];
  }
  if (idx < 8192) {                                    // wr1cP: 16 n-tiles x 512
    int e = idx & 15, lane = (idx >> 4) & 31, nt = idx >> 9;
    int k = ((e >> 3) << 4) + ((lane >> 4) << 3) + (e & 7);
    wr1cP[idx] = (_Float16)Wr1[(96 + k) * 256 + nt * 16 + (lane & 15)];
  }
}

// K1: metric = sym( reshape(relu(p@Wm1+bm1)@Wm2+bm2) + eps*I )
__global__ void k1_metric(const float* __restrict__ points,
                          const float* __restrict__ Wm1, const float* __restrict__ bm1,
                          const float* __restrict__ Wm2, const float* __restrict__ bm2,
                          float* __restrict__ metric) {
  __shared__ float pts[32], hid[128], comps[1024];
  int b = blockIdx.x, t = threadIdx.x;
  if (t < 32) pts[t] = points[b * 32 + t];
  __syncthreads();
  if (t < 128) {
    float a = bm1[t];
    for (int m = 0; m < 32; ++m) a += pts[m] * Wm1[m * 128 + t];
    hid[t] = fmaxf(a, 0.f);
  }
  __syncthreads();
  for (int n = t; n < 1024; n += 256) {
    float a = bm2[n];
    for (int h = 0; h < 128; ++h) a += hid[h] * Wm2[h * 1024 + n];
    if ((n >> 5) == (n & 31)) a += 1e-6f;
    comps[n] = a;
  }
  __syncthreads();
  for (int n = t; n < 1024; n += 256) {
    int i = n >> 5, j = n & 31;
    metric[b * 1024 + n] = 0.5f * (comps[n] + comps[j * 32 + i]);
  }
}

// K2: per-b precompute of Wr1 partial products; also zero hsum accumulator
__global__ void k2_pre(const float* __restrict__ points, const float* __restrict__ Wr1,
                       const float* __restrict__ metric,
                       float* __restrict__ ptsT, float* __restrict__ g1,
                       float* __restrict__ g2, float* __restrict__ hsum) {
  __shared__ float pts[32], met[1024];
  int b = blockIdx.x, t = threadIdx.x;
  if (t < 32) pts[t] = points[b * 32 + t];
  for (int n = t; n < 1024; n += 256) met[n] = metric[b * 1024 + n];
  hsum[b * 256 + t] = 0.f;
  __syncthreads();
  {
    float a = 0.f;
    for (int m = 0; m < 32; ++m) a += pts[m] * Wr1[m * 256 + t];
    ptsT[b * 256 + t] = a;
  }
  for (int r = 0; r < 32; ++r) {
    float a1 = 0.f, a2 = 0.f;
    for (int k = 0; k < 32; ++k) {
      float mv = met[r * 32 + k];
      a1 += mv * Wr1[(32 + k) * 256 + t];
      a2 += mv * Wr1[(64 + k) * 256 + t];
    }
    g1[(b * 32 + r) * 256 + t] = a1;
    g2[(b * 32 + r) * 256 + t] = a2;
  }
}

// K3: per (b,i): Christoffel MLP (VALU+tanh; K=3 so WMMA is wasteful there),
// chris written to LDS directly in f16 A-fragment layout, then
// cterm(32x256) = chris @ Wr1c via WMMA with contiguous b128 fragment loads,
// then relu-sum over j accumulated into hsum[b].
__global__ void __launch_bounds__(256)
k3_chris_ricci(const float* __restrict__ metric,
               const float* __restrict__ Wc1, const float* __restrict__ bc1,
               const float* __restrict__ Wc2, const float* __restrict__ bc2,
               const _Float16* __restrict__ wr1cP, const float* __restrict__ br1,
               const float* __restrict__ ptsT, const float* __restrict__ g1,
               const float* __restrict__ g2, float* __restrict__ hsum) {
  __shared__ float met[1024];
  __shared__ _Float16 chrisA[1024];       // 2 A-tiles x (32 lanes x 16 halfs)
  __shared__ float cterm[32 * 256];
  int blk = blockIdx.x;
  int b = blk >> 5, i = blk & 31;
  int t = threadIdx.x;
#if defined(__gfx1250__) && __has_builtin(__builtin_amdgcn_global_load_async_to_lds_b128) && __has_builtin(__builtin_amdgcn_s_wait_asynccnt)
  // one ASYNC b128 per lane: 256 lanes x 16B = whole 4KB metric tile
  __builtin_amdgcn_global_load_async_to_lds_b128(
      (__attribute__((address_space(1))) v4i*)(metric + b * 1024 + t * 4),
      (__attribute__((address_space(3))) v4i*)(met + t * 4), 0, 0);
  __builtin_amdgcn_s_wait_asynccnt(0);
#else
  for (int n = t; n < 1024; n += 256) met[n] = metric[b * 1024 + n];
#endif
  __syncthreads();
  float bc2v = bc2[0];
  for (int p = t; p < 1024; p += 256) {        // p = j*32 + k
    int j = p >> 5, k = p & 31;
    float gij = met[i * 32 + j], gjk = met[j * 32 + k], gki = met[k * 32 + i];
    float acc = bc2v;
    for (int n = 0; n < 128; ++n) {
      float h = tanhf(gij * Wc1[n] + gjk * Wc1[128 + n] + gki * Wc1[256 + n] + bc1[n]);
      acc += h * Wc2[n];
    }
    // store straight into A-fragment layout: row j, k-index k
    int lane2 = (j & 15) + (((k >> 3) & 1) << 4);
    int e2 = ((k >> 4) << 3) + (k & 7);
    chrisA[(j >> 4) * 512 + lane2 * 16 + e2] = (_Float16)acc;
  }
  __syncthreads();
  // WMMA: 2 m-tiles x 16 n-tiles, one v_wmma_f32_16x16x32_f16 per tile
  int wave = t >> 5, lane = t & 31;
  int lmod = lane & 15, lhi = lane >> 4;
  for (int q = 0; q < 4; ++q) {
    int tile = wave * 4 + q;          // 0..31
    int mt = tile >> 4, nt = tile & 15;
    v16h a  = *(const v16h*)&chrisA[mt * 512 + lane * 16];   // 2x ds_load_b128
    v16h bb = *(const v16h*)&wr1cP[nt * 512 + lane * 16];    // 2x global_load_b128
    v8f c = {};
    c = __builtin_amdgcn_wmma_f32_16x16x32_f16(false, a, false, bb, (short)0, c, false, false);
    for (int v = 0; v < 8; ++v) {
      int m = mt * 16 + v + (lhi << 3);                      // C/D: M = vgpr + 8*(lane/16)
      cterm[m * 256 + nt * 16 + lmod] = c[v];
    }
  }
  __syncthreads();
  {
    int n = t;  // 0..255
    float base = ptsT[b * 256 + n] + g1[(b * 32 + i) * 256 + n] + br1[n];
    float acc = 0.f;
    for (int j = 0; j < 32; ++j) {
      float h = base + g2[(b * 32 + j) * 256 + n] + cterm[j * 256 + n];
      acc += fmaxf(h, 0.f);
    }
    atomicAdd(&hsum[b * 256 + n], acc);
  }
}

// K4: S(32x1024) = hsum(32x256) @ Wr2 + 1024*br2; hsum shuffled once into
// f16 A-fragment layout in LDS, B pre-packed -> all fragment loads are b128.
__global__ void __launch_bounds__(256)
k4_sgemm(const float* __restrict__ hsum, const _Float16* __restrict__ wr2P,
         const float* __restrict__ br2, float* __restrict__ S) {
  __shared__ _Float16 hhA[8192];         // (mt*8+kt) tiles x 512
  int t = threadIdx.x;
  for (int n = t; n < 8192; n += 256) {
    int m = n >> 8, kk = n & 255;
    int mt = m >> 4, kt = kk >> 5, kl = kk & 31;
    int lane2 = (m & 15) + (((kl >> 3) & 1) << 4);
    int e2 = ((kl >> 4) << 3) + (kl & 7);
    hhA[(mt * 8 + kt) * 512 + lane2 * 16 + e2] = (_Float16)hsum[n];
  }
  __syncthreads();
  int wave = t >> 5, lane = t & 31;
  int lmod = lane & 15, lhi = lane >> 4;
  for (int nq = 0; nq < 8; ++nq) {
    int nt = wave * 8 + nq;              // 64 n-tiles total
    for (int mt = 0; mt < 2; ++mt) {
      v8f acc = {};
      for (int kt = 0; kt < 8; ++kt) {
        v16h a  = *(const v16h*)&hhA[(mt * 8 + kt) * 512 + lane * 16];
        v16h bb = *(const v16h*)&wr2P[((kt * 64 + nt) << 9) + lane * 16];
        acc = __builtin_amdgcn_wmma_f32_16x16x32_f16(false, a, false, bb, (short)0, acc, false, false);
      }
      int n = nt * 16 + lmod;
      for (int v = 0; v < 8; ++v) {
        int m = mt * 16 + v + (lhi << 3);
        S[m * 1024 + n] = acc[v] + 1024.f * br2[n];
      }
    }
  }
}

// K5: ricci = sym(S)/1024, ricci_dir, flow net, Hamiltonian net (tiny tail)
__global__ void k5_final(const float* __restrict__ points, const float* __restrict__ S,
                         const float* __restrict__ Wf1, const float* __restrict__ bf1,
                         const float* __restrict__ Wf2, const float* __restrict__ bf2,
                         const float* __restrict__ Wh1, const float* __restrict__ bh1,
                         const float* __restrict__ Wh2, const float* __restrict__ bh2,
                         float* __restrict__ out) {
  __shared__ float ricci[1024], pts[32], rdir[32], fh[128], npos[32], hhid[128];
  int b = blockIdx.x, t = threadIdx.x;
  if (t < 32) pts[t] = points[b * 32 + t];
  for (int n = t; n < 1024; n += 256) {
    int k = n >> 5, l = n & 31;
    ricci[n] = 0.5f * (S[b * 1024 + k * 32 + l] + S[b * 1024 + l * 32 + k]) * (1.f / 1024.f);
  }
  __syncthreads();
  if (t < 32) {
    float a = 0.f;
    for (int j = 0; j < 32; ++j) a += ricci[t * 32 + j] * pts[j];
    rdir[t] = a;
  }
  __syncthreads();
  if (t < 128) {
    float a = bf1[t];
    for (int m = 0; m < 32; ++m) a += pts[m] * Wf1[m * 128 + t] + rdir[m] * Wf1[(32 + m) * 128 + t];
    fh[t] = fmaxf(a, 0.f);
  }
  __syncthreads();
  if (t < 32) {
    float a = bf2[t];
    for (int h = 0; h < 128; ++h) a += fh[h] * Wf2[h * 32 + t];
    npos[t] = pts[t] + a;
  }
  __syncthreads();
  if (t < 128) {                      // phase upper half is zero -> only Wh1 rows 0..31
    float a = bh1[t];
    for (int m = 0; m < 32; ++m) a += npos[m] * Wh1[m * 128 + t];
    hhid[t] = tanhf(a);
  }
  __syncthreads();
  if (t < 64) {
    float a = bh2[t];
    for (int h = 0; h < 128; ++h) a += hhid[h] * Wh2[h * 64 + t];
    out[b * 64 + t] = a;
  }
}

extern "C" void kernel_launch(void* const* d_in, const int* in_sizes, int n_in,
                              void* d_out, int out_size, void* d_ws, size_t ws_size,
                              hipStream_t stream) {
  (void)in_sizes; (void)n_in; (void)out_size; (void)ws_size;
  const float* points = (const float*)d_in[0];
  const float* Wm1 = (const float*)d_in[1];  const float* bm1 = (const float*)d_in[2];
  const float* Wm2 = (const float*)d_in[3];  const float* bm2 = (const float*)d_in[4];
  const float* Wc1 = (const float*)d_in[5];  const float* bc1 = (const float*)d_in[6];
  const float* Wc2 = (const float*)d_in[7];  const float* bc2 = (const float*)d_in[8];
  const float* Wr1 = (const float*)d_in[9];  const float* br1 = (const float*)d_in[10];
  const float* Wr2 = (const float*)d_in[11]; const float* br2 = (const float*)d_in[12];
  const float* Wf1 = (const float*)d_in[13]; const float* bf1 = (const float*)d_in[14];
  const float* Wf2 = (const float*)d_in[15]; const float* bf2 = (const float*)d_in[16];
  const float* Wh1 = (const float*)d_in[17]; const float* bh1 = (const float*)d_in[18];
  const float* Wh2 = (const float*)d_in[19]; const float* bh2 = (const float*)d_in[20];

  float* ws     = (float*)d_ws;
  float* metric = ws + OFF_METRIC;
  float* ptsT   = ws + OFF_PTST;
  float* g1     = ws + OFF_G1;
  float* g2     = ws + OFF_G2;
  float* hsum   = ws + OFF_HSUM;
  float* S      = ws + OFF_S;
  _Float16* wr1cP = (_Float16*)(ws + F32_WS_FLOATS);   // 8192 halfs, 32B aligned
  _Float16* wr2P  = wr1cP + 8192;                       // 262144 halfs, 32B aligned
  float* out = (float*)d_out;

  hipLaunchKernelGGL(k0_pack, dim3(1024), dim3(256), 0, stream, Wr1, Wr2, wr1cP, wr2P);
  hipLaunchKernelGGL(k1_metric, dim3(32), dim3(256), 0, stream,
                     points, Wm1, bm1, Wm2, bm2, metric);
  hipLaunchKernelGGL(k2_pre, dim3(32), dim3(256), 0, stream,
                     points, Wr1, metric, ptsT, g1, g2, hsum);
  hipLaunchKernelGGL(k3_chris_ricci, dim3(1024), dim3(256), 0, stream,
                     metric, Wc1, bc1, Wc2, bc2, wr1cP, br1, ptsT, g1, g2, hsum);
  hipLaunchKernelGGL(k4_sgemm, dim3(1), dim3(256), 0, stream, hsum, wr2P, br2, S);
  hipLaunchKernelGGL(k5_final, dim3(32), dim3(256), 0, stream,
                     points, S, Wf1, bf1, Wf2, bf2, Wh1, bh1, Wh2, bh2, out);
}
